// UltimateGeoSNN_11012296147135
// MI455X (gfx1250) — compile-verified, matrix operand order checked by reading, not run on hardware
//
#include <hip/hip_runtime.h>

// ============================================================================
// MI455X (gfx1250) fused SNN forward.
//
// Roofline: input x = 157 MB, math ~30 GFLOP -> memory bound (6.7us HBM floor
// at 23.3 TB/s). Design: 1 workgroup per batch row, entire pipeline LDS
// resident (~155KB of 320KB/WGP), HBM traffic = one read of x + (B,4) output.
// GEMM-shaped stages use V_WMMA_F32_16X16X32_F16 (f32 accumulate); input
// staging uses the Tensor Data Mover (tensor_load_to_lds / s_wait_tensorcnt).
// ============================================================================

typedef _Float16 half_t;
typedef __attribute__((ext_vector_type(16))) _Float16 v16h;
typedef __attribute__((ext_vector_type(2)))  _Float16 h2;
typedef __attribute__((ext_vector_type(8)))  float    v8f;
typedef __attribute__((ext_vector_type(4)))  float    f32x4;
typedef __attribute__((ext_vector_type(4)))  unsigned u32x4;
typedef __attribute__((ext_vector_type(8)))  int      i32x8;
typedef __attribute__((ext_vector_type(4)))  int      i32x4;

#define EPS 1e-5f
#define NBATCH 2048
#define NT     320

// ---- workspace layout (bytes) ----------------------------------------------
#define O_MA   0        // stage-A B-fragments: 2 slices x 32 lanes x 16 halves
#define O_SA   2048     // stage-A BN scale  [16] f32
#define O_OA   2112     // stage-A BN offset [16] f32
#define O_WSP  2176     // folded 1x1 spatial weights [16][16] f32
#define O_CSP  3200     // spatial const [16] f32
#define O_W1   3264     // conv1 B-frags: 2 tiles x 8 slices x 32 x 16 halves
#define O_C1   19648    // conv1 const [32] f32
#define O_W2   19776    // conv2 B-frags: 4 tiles x 7 slices x 32 x 16 halves
#define O_C2   48448    // conv2 const [64] f32
#define O_FCW  48704    // folded fc weights [4][640] f32
#define O_FCB  58944    // folded fc bias [4] f32
#define O_DEC  58960    // decays sigmoid(w_v2), sigmoid(w_v3)
#define WS_BYTES 58968

// ---- LDS layout (bytes), regions overlaid by phase liveness ----------------
#define LDS_X    0       // x[b]  [15][320][4] f32  (phase A only)
#define LDS_SSP  0       // s_spatial [320][16] f16 (post phase A)
#define LDS_C1   10240   // c1 [320][32] f32
#define LDS_S2   51200   // s2 spikes [320][32] f16
#define LDS_POOL 71680   // pooled [10][64] f32
#define LDS_O4   74240   // fc accum [4] f32
#define LDS_INJ  76800   // inj [320][16] f32  (dead before conv2)
#define LDS_C2   76800   // c2 [320][64] f32   (reuses inj/s1 region)
#define LDS_S1   97280   // s1 spikes [320][16] f16
#define LDS_BYTES 158720

struct PtrPack { const float* p[44]; };

// K-index within a 16x16x32 f16 fragment, per ISA 7.12.2 16-bit A layout:
// lanes 0-15: VGPR0..3 hold K=0..7 pairs, VGPR4..7 hold K=16..23;
// lanes 16-31: +8.  (B assumed mirrored with N striped across lanes.)
__device__ __forceinline__ int kbase_of(int v, int h, int hilane) {
    int kb = (v < 4) ? (2 * v + h) : (16 + 2 * (v - 4) + h);
    return kb + (hilane ? 8 : 0);
}

__device__ __forceinline__ v8f wmma_f16(v16h a, v16h b, v8f c) {
    return __builtin_amdgcn_wmma_f32_16x16x32_f16(false, a, false, b,
                                                  (short)0, c, false, false);
}

// ============================================================================
// Prep kernel: fold BN into weights, emit WMMA B-fragments into workspace.
// Runs once per launch; cost negligible vs. the 2048-WG main kernel.
// ============================================================================
__global__ __launch_bounds__(256) void snn_prep(PtrPack P, float* ws) {
    const int tid = threadIdx.x;
    char* wsb = (char*)ws;
    half_t* frA  = (half_t*)(wsb + O_MA);
    float*  sA   = (float*)(wsb + O_SA);
    float*  oA   = (float*)(wsb + O_OA);
    float*  wsp  = (float*)(wsb + O_WSP);
    float*  csp  = (float*)(wsb + O_CSP);
    half_t* frW1 = (half_t*)(wsb + O_W1);
    float*  cst1 = (float*)(wsb + O_C1);
    half_t* frW2 = (half_t*)(wsb + O_W2);
    float*  cst2 = (float*)(wsb + O_C2);
    float*  fcW  = (float*)(wsb + O_FCW);
    float*  fcB  = (float*)(wsb + O_FCB);
    float*  dec  = (float*)(wsb + O_DEC);

    // ---- stage A: Mcomb[kappa=k*16+c][e] = (A_norm @ W_k^T)[c][e] ----------
    for (int i = tid; i < 1024; i += 256) {
        int s  = i >> 9;
        int ln = (i >> 4) & 31;
        int el = i & 15;
        int v = el >> 1, h = el & 1;
        int kk = 32 * s + kbase_of(v, h, ln >= 16);
        int k = kk >> 4, c = kk & 15;
        int n = ln & 15;
        float val = 0.f;
        if (c < 15 && n < 15) {
            const float* A = P.p[1];
            const float* W = P.p[2 + 2 * k];  // W_u, W_v, W_c, W_a
            for (int d = 0; d < 15; ++d) val += A[c * 15 + d] * W[n * 15 + d];
        }
        frA[i] = (half_t)val;
    }
    // ---- stage A BN affine (+ summed per-gate biases) -----------------------
    if (tid < 16) {
        float sc = 0.f, off = 0.f;
        if (tid < 15) {
            sc = P.p[10][tid] * rsqrtf(P.p[13][tid] + EPS);
            float bias = P.p[3][tid] + P.p[5][tid] + P.p[7][tid] + P.p[9][tid];
            off = (bias - P.p[12][tid]) * sc + P.p[11][tid];
        }
        sA[tid] = sc; oA[tid] = off;
    }
    // ---- spatial 1x1 conv + BN folded ---------------------------------------
    if (tid < 16) {
        float sc = P.p[16][tid] * rsqrtf(P.p[19][tid] + EPS);
        csp[tid] = (P.p[15][tid] - P.p[18][tid]) * sc + P.p[17][tid];
        for (int c = 0; c < 16; ++c)
            wsp[tid * 16 + c] = (c < 15) ? P.p[14][tid * 15 + c] * sc : 0.f;
    }
    // ---- conv1 consts: BN + shortcut bias -----------------------------------
    if (tid < 32) {
        float sc = P.p[22][tid] * rsqrtf(P.p[25][tid] + EPS);
        cst1[tid] = (P.p[21][tid] - P.p[24][tid]) * sc + P.p[23][tid] + P.p[27][tid];
    }
    // ---- conv1 fragments: kappa = r*16+c, K=240 padded to 256 ---------------
    for (int i = tid; i < 8192; i += 256) {
        int et = i >> 12;
        int s  = (i >> 9) & 7;
        int ln = (i >> 4) & 31;
        int el = i & 15;
        int kk = 32 * s + kbase_of(el >> 1, el & 1, ln >= 16);
        int r = kk >> 4, c = kk & 15;
        int e = et * 16 + (ln & 15);
        float val = 0.f;
        if (r < 15) {
            float sc = P.p[22][e] * rsqrtf(P.p[25][e] + EPS);
            val = P.p[20][(e * 16 + c) * 15 + r] * sc;      // BN-scaled conv1_w
            if (r == 7) val += P.p[26][e * 16 + c];         // shortcut into center tap
        }
        frW1[i] = (half_t)val;
    }
    // ---- conv2 consts -------------------------------------------------------
    if (tid < 64) {
        float sc = P.p[30][tid] * rsqrtf(P.p[33][tid] + EPS);
        cst2[tid] = (P.p[29][tid] - P.p[32][tid]) * sc + P.p[31][tid] + P.p[35][tid];
    }
    // ---- conv2 fragments: kappa = r*32+c, K=224 exactly 7 slices ------------
    for (int i = tid; i < 14336; i += 256) {
        int et  = i / 3584;
        int rem = i % 3584;
        int s   = rem >> 9;
        int ln  = (rem >> 4) & 31;
        int el  = i & 15;
        int kk = 32 * s + kbase_of(el >> 1, el & 1, ln >= 16);
        int r = kk >> 5, c = kk & 31;
        int e = et * 16 + (ln & 15);
        float sc = P.p[30][e] * rsqrtf(P.p[33][e] + EPS);
        float val = P.p[28][(e * 32 + c) * 7 + r] * sc;
        if (r == 3) val += P.p[34][e * 32 + c];
        frW2[i] = (half_t)val;
    }
    // ---- FC with pre_fc_bn folded -------------------------------------------
    for (int i = tid; i < 2560; i += 256) {
        int ii = i % 640;
        float s = P.p[36][ii] * rsqrtf(P.p[39][ii] + EPS);
        fcW[i] = P.p[40][i] * s;
    }
    if (tid < 4) {
        float acc = P.p[41][tid];
        for (int ii = 0; ii < 640; ++ii) {
            float s = P.p[36][ii] * rsqrtf(P.p[39][ii] + EPS);
            acc += (P.p[37][ii] - P.p[38][ii] * s) * P.p[40][tid * 640 + ii];
        }
        fcB[tid] = acc;
    }
    if (tid == 0) {
        dec[0] = 1.f / (1.f + expf(-P.p[42][0]));  // sigmoid(w_v2)
        dec[1] = 1.f / (1.f + expf(-P.p[43][0]));  // sigmoid(w_v3)
    }
}

// ============================================================================
// Main fused kernel: one workgroup (8 wave32) per batch row.
// ============================================================================
__global__ __launch_bounds__(256) void snn_main(const float* __restrict__ x,
                                                const float* __restrict__ ws,
                                                float* __restrict__ out) {
    extern __shared__ char smem[];
    const int b    = blockIdx.x;
    const int tid  = threadIdx.x;
    const int wave = tid >> 5;
    const int lane = tid & 31;
    const int lh   = lane & 15;
    const int hi   = lane >= 16;

    float*  xb   = (float*)(smem + LDS_X);
    float*  inj  = (float*)(smem + LDS_INJ);
    half_t* s1   = (half_t*)(smem + LDS_S1);
    half_t* ssp  = (half_t*)(smem + LDS_SSP);
    float*  c1   = (float*)(smem + LDS_C1);
    half_t* s2   = (half_t*)(smem + LDS_S2);
    float*  c2   = (float*)(smem + LDS_C2);
    float*  pool = (float*)(smem + LDS_POOL);
    float*  o4   = (float*)(smem + LDS_O4);

    const char* wsb = (const char*)ws;
    const half_t* frA  = (const half_t*)(wsb + O_MA);
    const float*  sA   = (const float*)(wsb + O_SA);
    const float*  oA   = (const float*)(wsb + O_OA);
    const float*  wspW = (const float*)(wsb + O_WSP);
    const float*  cspC = (const float*)(wsb + O_CSP);
    const half_t* frW1 = (const half_t*)(wsb + O_W1);
    const float*  cst1 = (const float*)(wsb + O_C1);
    const half_t* frW2 = (const half_t*)(wsb + O_W2);
    const float*  cst2 = (const float*)(wsb + O_C2);
    const float*  fcW  = (const float*)(wsb + O_FCW);
    const float*  fcB  = (const float*)(wsb + O_FCB);
    const float*  dec  = (const float*)(wsb + O_DEC);

    const float* xg = x + (size_t)b * 19200;   // x[b] : [15][320][4] f32
    __builtin_prefetch(frW1, 0, 1);            // global_prefetch_b8 (weights hot in L2)
    __builtin_prefetch(frW2, 0, 1);

    // ------------------------------------------------------------------
    // Phase 0: stage x[b] (76.8KB, contiguous) into LDS via Tensor Data
    // Mover (1D tile covering the whole slab). Fallback: vector copy.
    // ------------------------------------------------------------------
#if defined(__gfx1250__) && __has_builtin(__builtin_amdgcn_tensor_load_to_lds)
    if (wave == 0) {
        unsigned long long ga = (unsigned long long)xg;
        u32x4 g0;
        g0[0] = 1u;                                      // count=1, user descriptor
        g0[1] = (unsigned)LDS_X;                         // lds_addr
        g0[2] = (unsigned)(ga & 0xFFFFFFFFull);          // global_addr lo
        g0[3] = (unsigned)((ga >> 32) & 0x01FFFFFFull) | 0x80000000u; // hi | type=2
        i32x8 g1;
        g1[0] = (int)(2u << 16);        // data_size = 4B; workgroup_mask = 0
        g1[1] = (int)(19200u << 16);    // tensor_dim0[15:0]
        g1[2] = (int)(1u << 16);        // tensor_dim0 hi = 0; tensor_dim1 = 1
        g1[3] = (int)(19200u << 16);    // tensor_dim1 hi = 0; tile_dim0 = 19200
        g1[4] = 1;                      // tile_dim1 = 1; tile_dim2 = 0
        g1[5] = 19200;                  // tensor_dim0_stride lo
        g1[6] = 0;
        g1[7] = 0;
        i32x4 gz = {0, 0, 0, 0};
#if __clang_major__ >= 23
        i32x8 gz8 = {0, 0, 0, 0, 0, 0, 0, 0};
        __builtin_amdgcn_tensor_load_to_lds(g0, g1, gz, gz, gz8, 0);
#else
        __builtin_amdgcn_tensor_load_to_lds(g0, g1, gz, gz, 0);
#endif
#if __has_builtin(__builtin_amdgcn_s_wait_tensorcnt)
        __builtin_amdgcn_s_wait_tensorcnt(0);
#else
        asm volatile("s_wait_tensorcnt 0x0" ::: "memory");
#endif
    }
#else
    for (int i = tid; i < 4800; i += 256)
        ((f32x4*)xb)[i] = ((const f32x4*)xg)[i];
#endif
    __syncthreads();

    // ------------------------------------------------------------------
    // Phase A: inj[t][e] = sum_{k,c} x[c][t][k] * Mcomb[k*16+c][e], BN folded.
    // GEMM tiles: D(16t x 16e) = A(16 x 64) x B(64 x 16), 2 K-slices of 32.
    // ------------------------------------------------------------------
    for (int tt = wave; tt < 20; tt += 8) {
        v8f acc = {};
        const int t = tt * 16 + lh;
        for (int s = 0; s < 2; ++s) {
            v16h a;
#pragma unroll
            for (int el = 0; el < 16; ++el) {
                int kk = 32 * s + kbase_of(el >> 1, el & 1, hi);
                int k = kk >> 4, c = kk & 15;
                float xv = (c < 15) ? xb[(c * 320 + t) * 4 + k] : 0.f;
                a[el] = (half_t)xv;
            }
            v16h bf = *(const v16h*)(frA + (s * 32 + lane) * 16);
            acc = wmma_f16(a, bf, acc);
        }
#pragma unroll
        for (int r = 0; r < 8; ++r) {
            int m = r + (hi ? 8 : 0);
            inj[(tt * 16 + m) * 16 + lh] = acc[r] * sA[lh] + oA[lh];
        }
    }
    __syncthreads();

    // ---- LIF 1: decay exp(-1/35), v_th = 3 ----------------------------------
    if (tid < 32) {
        const float d1 = expf(-1.f / 35.f);
        float v = 0.f;
        const int c = lane;
        for (int t = 0; t < 320; ++t) {
            float sp = 0.f;
            if (c < 15) {
                v = v * d1 + inj[t * 16 + c];
                sp = (v > 3.f) ? 1.f : 0.f;
                v -= sp * 3.f;
            }
            if (c < 16) s1[t * 16 + c] = (half_t)sp;   // c==15 pads zero
        }
    }
    __syncthreads();

    // ---- spatial 1x1 conv (15->16) + BN, folded -----------------------------
    for (int i = tid; i < 5120; i += 256) {
        int t = i >> 4, e = i & 15;
        float acc = cspC[e];
        for (int c = 0; c < 15; ++c) acc += wspW[e * 16 + c] * (float)s1[t * 16 + c];
        ssp[t * 16 + e] = (half_t)acc;
    }
    __syncthreads();

    // ------------------------------------------------------------------
    // conv1 (16->32, k=15, pad 7) + BN + shortcut, all folded into one
    // im2col GEMM: K = r*16+c (240 -> 8 slices). 40 output tiles.
    // ------------------------------------------------------------------
    for (int idx = wave; idx < 40; idx += 8) {
        int tt = idx >> 1, et = idx & 1;
        v8f acc = {};
        const int t = tt * 16 + lh;
        for (int s = 0; s < 8; ++s) {
            v16h a;
#pragma unroll
            for (int p = 0; p < 8; ++p) {
                int kk = 32 * s + kbase_of(p, 0, hi);   // even -> pair (kk, kk+1)
                int r = kk >> 4, c = kk & 15;
                int ts = t + r - 7;
                h2 pr = {};
                if (r < 15 && ts >= 0 && ts < 320)
                    pr = *(const h2*)&ssp[ts * 16 + c];
                a[2 * p] = pr[0]; a[2 * p + 1] = pr[1];
            }
            v16h bf = *(const v16h*)(frW1 + ((et * 8 + s) * 32 + lane) * 16);
            acc = wmma_f16(a, bf, acc);
        }
#pragma unroll
        for (int r = 0; r < 8; ++r) {
            int m = r + (hi ? 8 : 0);
            int e = et * 16 + lh;
            c1[(tt * 16 + m) * 32 + e] = acc[r] + cst1[e];
        }
    }
    __syncthreads();

    // ---- LIF 2: decay sigmoid(w_v2), v_th = 2 -------------------------------
    if (tid < 32) {
        const float d2 = dec[0];
        float v = 0.f;
        for (int t = 0; t < 320; ++t) {
            v = v * d2 + c1[t * 32 + lane];
            float sp = (v > 2.f) ? 1.f : 0.f;
            v -= sp * 2.f;
            s2[t * 32 + lane] = (half_t)sp;
        }
    }
    __syncthreads();

    // ------------------------------------------------------------------
    // conv2 (32->64, k=7, pad 3) + BN + shortcut: K = r*32+c = 224 -> 7 slices.
    // 80 output tiles.
    // ------------------------------------------------------------------
    for (int idx = wave; idx < 80; idx += 8) {
        int tt = idx >> 2, et = idx & 3;
        v8f acc = {};
        const int t = tt * 16 + lh;
        for (int s = 0; s < 7; ++s) {
            v16h a;
#pragma unroll
            for (int p = 0; p < 8; ++p) {
                int kk = 32 * s + kbase_of(p, 0, hi);
                int r = kk >> 5, c = kk & 31;
                int ts = t + r - 3;
                h2 pr = {};
                if (ts >= 0 && ts < 320)
                    pr = *(const h2*)&s2[ts * 32 + c];
                a[2 * p] = pr[0]; a[2 * p + 1] = pr[1];
            }
            v16h bf = *(const v16h*)(frW2 + ((et * 7 + s) * 32 + lane) * 16);
            acc = wmma_f16(a, bf, acc);
        }
#pragma unroll
        for (int r = 0; r < 8; ++r) {
            int m = r + (hi ? 8 : 0);
            int e = et * 16 + lh;
            c2[(tt * 16 + m) * 64 + e] = acc[r] + cst2[e];
        }
    }
    __syncthreads();

    // ---- LIF 3 (decay sigmoid(w_v3), v_th=1.5) fused with diff-pooling ------
    if (tid < 64) {
        const float d3 = dec[1];
        float v = 0.f, wacc = 0.f;
        const int e = tid;
        for (int t = 0; t < 320; ++t) {
            v = v * d3 + c2[t * 64 + e];
            float sp = (v > 1.5f) ? 1.f : 0.f;
            v -= sp * 1.5f;
            wacc += ((t & 31) < 16) ? -sp : sp;        // mean(last16)-mean(first16)
            if ((t & 31) == 31) { pool[(t >> 5) * 64 + e] = wacc * (1.f / 16.f); wacc = 0.f; }
        }
    }
    if (tid == 0) { o4[0] = 0.f; o4[1] = 0.f; o4[2] = 0.f; o4[3] = 0.f; }
    __syncthreads();

    // ---- FC 640 -> 4 (pre_fc_bn folded); reduce via LDS float atomics -------
    {
        float pj[4] = {0.f, 0.f, 0.f, 0.f};
        for (int i = tid; i < 640; i += 256) {
            int e = i / 10, w = i % 10;                 // flat index = e*10 + w
            float pv = pool[w * 64 + e];
#pragma unroll
            for (int j = 0; j < 4; ++j) pj[j] += pv * fcW[j * 640 + i];
        }
#pragma unroll
        for (int j = 0; j < 4; ++j) atomicAdd(&o4[j], pj[j]);
    }
    __syncthreads();
    if (tid < 4) out[b * 4 + tid] = o4[tid] + fcB[tid];
}

// ============================================================================
extern "C" void kernel_launch(void* const* d_in, const int* in_sizes, int n_in,
                              void* d_out, int out_size, void* d_ws, size_t ws_size,
                              hipStream_t stream) {
    (void)in_sizes; (void)out_size; (void)ws_size;
    PtrPack P;
    for (int i = 0; i < 44; ++i) P.p[i] = (i < n_in) ? (const float*)d_in[i] : nullptr;
    float* ws = (float*)d_ws;

    hipFuncSetAttribute(reinterpret_cast<const void*>(snn_main),
                        hipFuncAttributeMaxDynamicSharedMemorySize, LDS_BYTES);

    snn_prep<<<1, 256, 0, stream>>>(P, ws);
    snn_main<<<NBATCH, 256, LDS_BYTES, stream>>>((const float*)d_in[0], ws,
                                                 (float*)d_out);
}
